// ToepliztV2_70935679860760
// MI455X (gfx1250) — matrix operand order repaired and checked by reading, not compiled, not run on hardware
//
#include <hip/hip_runtime.h>
#include <stdint.h>

// out[b,t,d] = sum_s a[(t-s) mod 2n] * x[b,s,d]   (Toeplitz matvec == FFT conv)
// Per-batch GEMM: T(4096x4096, from 8192-vec) @ X(4096x512), bf16 WMMA, f32 acc.
// (exp <= e^30 ~ 1e13 overflows f16; bf16 keeps f32 range.)
//
// d_ws layout:  [0, 512KB)        A-fragment table (512 frags x 512 bf16, WMMA A layout)
//               [512KB, 512KB+32MB) xT[b][d][k] bf16 (transposed, converted x)

#define B_    8
#define N_    4096
#define D_    512
#define TWO_N 8192

#define BT 128   // t-rows per workgroup
#define BD 256   // d-cols per workgroup
#define KT 32    // k-chunk (WMMA bf16 K)

#define FRAG_BYTES  (512u * 512u * 2u)   // 512 frags * 512 bf16
#define XT_OFF      ((size_t)FRAG_BYTES) // 512 KB

typedef __attribute__((ext_vector_type(16))) __bf16   v16bf;
typedef __attribute__((ext_vector_type(8)))  float    v8f;
typedef __attribute__((ext_vector_type(4)))  uint32_t v4u;
typedef __attribute__((ext_vector_type(8)))  int      v8i;
typedef __attribute__((ext_vector_type(4)))  int      v4i;

__device__ __forceinline__ float toeplitz_a(int aidx, const float* pos,
                                            const float* zero, const float* neg) {
  float v;
  if (aidx == 0 || aidx == N_) v = zero[0];
  else if (aidx < N_)          v = pos[aidx - 1];
  else                         v = neg[aidx - N_ - 1];
  v = fminf(fmaxf(v, -60.0f), 30.0f);
  return expf(v);
}

// ---------------------------------------------------------------------------
// Prep 1: all possible WMMA A-fragments, indexed by id = (t_base-k0)/16 + 256.
// fragT[id*512 + lane*16 + e] = a[((id-256)*16 + m - kk(lane,e)) & 8191]
// kk mapping per ISA 7.12.2 (16-bit A-matrix 16x32 lane layout).
// ---------------------------------------------------------------------------
__global__ void frag_prep_kernel(const float* __restrict__ pos,
                                 const float* __restrict__ zero,
                                 const float* __restrict__ neg,
                                 __bf16* __restrict__ fragT) {
  int idx  = blockIdx.x * blockDim.x + threadIdx.x;  // [0, 512*512)
  int id   = idx >> 9;
  int rem  = idx & 511;
  int lane = rem >> 4;
  int e    = rem & 15;
  int m    = lane & 15;
  int hi   = (lane >= 16);
  int vv = e >> 1, h = e & 1;
  int kk = (vv < 4) ? ((hi ? 8 : 0) + 2 * vv + h)
                    : ((hi ? 24 : 16) + 2 * (vv - 4) + h);
  int diff = (id - 256) << 4;
  int aidx = (diff + m - kk) & (TWO_N - 1);
  fragT[idx] = (__bf16)toeplitz_a(aidx, pos, zero, neg);
}

// ---------------------------------------------------------------------------
// Prep 2: xT[b][d][k] = bf16(x[b][k][d])  (LDS tile transpose, coalesced both ways)
// ---------------------------------------------------------------------------
__global__ void xpose_kernel(const float* __restrict__ x, __bf16* __restrict__ xT) {
  __shared__ float tile[32][33];
  const int b = blockIdx.z, dt = blockIdx.x * 32, kt = blockIdx.y * 32;
  const int tx = threadIdx.x, ty = threadIdx.y;   // 32 x 8
  const float* xb  = x  + (size_t)b * N_ * D_;
  __bf16*      xTb = xT + (size_t)b * D_ * N_;
#pragma unroll
  for (int j = 0; j < 4; ++j) {
    int row = ty + j * 8;
    tile[row][tx] = xb[(size_t)(kt + row) * D_ + dt + tx];
  }
  __syncthreads();
#pragma unroll
  for (int j = 0; j < 4; ++j) {
    int row = ty + j * 8;
    xTb[(size_t)(dt + row) * N_ + kt + tx] = (__bf16)tile[tx][row];
  }
}

// ---------------------------------------------------------------------------
// Main GEMM: A-frags = 2x global b128 from frag table (L2-resident),
// B tile = TDM (tensor_load_to_lds) into double-buffered LDS [d][k].
// ---------------------------------------------------------------------------
__global__ __launch_bounds__(256) void toeplitz_wmma_kernel(
    const __bf16* __restrict__ xT,
    const __bf16* __restrict__ fragT,
    float* __restrict__ out) {
  __shared__ __align__(16) __bf16 lds_b0[BD * KT];   // 16 KB
  __shared__ __align__(16) __bf16 lds_b1[BD * KT];   // 16 KB

  const int tid  = threadIdx.x;
  const int lane = tid & 31;
  const int w    = tid >> 5;
  const int m    = lane & 15;
  const bool hi  = lane >= 16;

  const int bIdx  = blockIdx.z;
  const int t0    = blockIdx.y * BT;
  const int d0    = blockIdx.x * BD;
  const int waveT = (w >> 2) * 64;
  const int waveD = (w & 3) * 64;

  const __bf16* xTb = xT  + (size_t)bIdx * D_ * N_;
  float*        ob  = out + (size_t)bIdx * N_ * D_;

  // ---- B-tile staging: TDM when available, cooperative copy fallback ----
  auto stage_issue = [&](__bf16* buf, int k0) {
#if __has_builtin(__builtin_amdgcn_tensor_load_to_lds)
    if (w == 0) {
      const uint32_t lds_off = (uint32_t)(uintptr_t)buf;
      const uint64_t ga = (uint64_t)(uintptr_t)(xTb + (size_t)d0 * N_ + k0);
      v4u g0 = { 1u,                                    // count=1 (valid user D#)
                 lds_off,                               // lds_addr
                 (uint32_t)ga,                          // global_addr[31:0]
                 (((uint32_t)(ga >> 32)) & 0x01FFFFFFu) | (2u << 30) }; // [56:32]|type=2
      v8i g1;
      g1[0] = (int)(1u << 16);                          // data_size = 2 bytes
      g1[1] = (int)(((uint32_t)N_ & 0xFFFFu) << 16);    // tensor_dim0 lo16 (=4096)
      g1[2] = (int)(((uint32_t)N_ >> 16) | ((uint32_t)D_ << 16)); // dim0 hi | dim1 lo
      g1[3] = (int)((uint32_t)KT << 16);                // dim1 hi(0) | tile_dim0 = 32
      g1[4] = BD;                                       // tile_dim1 = 256, tile_dim2 = 0
      g1[5] = N_;                                       // tensor_dim0_stride lo32 = 4096
      g1[6] = 0;
      g1[7] = 0;
      v4i gz = {0, 0, 0, 0};
#if __has_include(<hip/amd_detail/amd_gfx1250_TDM.h>)
      v8i gz8 = {0, 0, 0, 0, 0, 0, 0, 0};
      __builtin_amdgcn_tensor_load_to_lds(g0, g1, gz, gz, gz8, 0);
#else
      __builtin_amdgcn_tensor_load_to_lds(g0, g1, gz, gz, 0);
#endif
    }
#else
    const __bf16* src = xTb + (size_t)d0 * N_ + k0;
    for (int row = tid; row < BD; row += 256) {
      const int4* s = (const int4*)(src + (size_t)row * N_);
      int4* d = (int4*)(buf + row * KT);
      d[0] = s[0]; d[1] = s[1]; d[2] = s[2]; d[3] = s[3];
    }
#endif
  };
  auto stage_wait = [&]() {
#if __has_builtin(__builtin_amdgcn_tensor_load_to_lds)
#if __has_builtin(__builtin_amdgcn_s_wait_tensorcnt)
    if (w == 0) __builtin_amdgcn_s_wait_tensorcnt(0);
#else
    if (w == 0) asm volatile("s_wait_tensorcnt 0x0" ::: "memory");
#endif
#endif
  };

  v8f acc[4][4];
  const v8f z8 = {0.f, 0.f, 0.f, 0.f, 0.f, 0.f, 0.f, 0.f};
#pragma unroll
  for (int r = 0; r < 4; ++r)
#pragma unroll
    for (int c = 0; c < 4; ++c) acc[r][c] = z8;

  auto step = [&](const __bf16* cur, __bf16* nxt, int k0, bool hasNext) {
    stage_wait();          // TDM filling `cur` complete (wave 0)
    __syncthreads();       // `cur` ready for all; all done reading `nxt`'s old data
    if (hasNext) stage_issue(nxt, k0 + KT);

    // A fragments: two b128 loads each from the precomputed table (L2-resident).
    v16bf afrag[4];
#pragma unroll
    for (int r = 0; r < 4; ++r) {
      const int id = ((t0 + waveT + r * 16 - k0) >> 4) + 256;   // in [2, 511]
      const int4* ap = (const int4*)(fragT + ((size_t)id << 9) + lane * 16);
      int4* dst = (int4*)&afrag[r];
      dst[0] = ap[0];
      dst[1] = ap[1];
      if (r == 0)  // prefetch next k-step's fragment line (id decreases by 2)
        __builtin_prefetch(fragT + ((size_t)(id - 2) << 9) + lane * 16, 0, 3);
    }
    // B fragments: two ds_load_b128 each from [d][k] LDS tile.
    v16bf bfrag[4];
#pragma unroll
    for (int c = 0; c < 4; ++c) {
      const int dd = waveD + c * 16 + m;
      const int4* sp = (const int4*)(cur + dd * KT + (hi ? 8 : 0));
      int4* dst = (int4*)&bfrag[c];
      dst[0] = sp[0];
      dst[1] = sp[2];
    }
#pragma unroll
    for (int r = 0; r < 4; ++r)
#pragma unroll
      for (int c = 0; c < 4; ++c)
        acc[r][c] = __builtin_amdgcn_wmma_f32_16x16x32_bf16(
            false, afrag[r], false, bfrag[c], (short)0, acc[r][c], false, false);
  };

  stage_issue(lds_b0, 0);
  for (int k0 = 0; k0 < N_; k0 += 2 * KT) {
    step(lds_b0, lds_b1, k0, true);
    step(lds_b1, lds_b0, k0 + KT, (k0 + 2 * KT) < N_);
  }

  // ---- epilogue: 16x16 f32 C layout: vgpr i -> M = i (+8 for lanes 16..31), N = lane&15
#pragma unroll
  for (int r = 0; r < 4; ++r) {
    const int trow = t0 + waveT + r * 16 + (hi ? 8 : 0);
#pragma unroll
    for (int c = 0; c < 4; ++c) {
      const int dcol = d0 + waveD + c * 16 + m;
#pragma unroll
      for (int i = 0; i < 8; ++i)
        ob[(size_t)(trow + i) * D_ + dcol] = acc[r][c][i];
    }
  }
}

// ---------------------------------------------------------------------------
extern "C" void kernel_launch(void* const* d_in, const int* in_sizes, int n_in,
                              void* d_out, int out_size, void* d_ws, size_t ws_size,
                              hipStream_t stream) {
  (void)in_sizes; (void)n_in; (void)out_size; (void)ws_size;  // needs ~33 MB ws
  const float* x    = (const float*)d_in[0];
  const float* pos  = (const float*)d_in[1];
  const float* zero = (const float*)d_in[2];
  const float* neg  = (const float*)d_in[3];
  float* out = (float*)d_out;

  __bf16* fragT = (__bf16*)d_ws;
  __bf16* xT    = (__bf16*)((char*)d_ws + XT_OFF);

  frag_prep_kernel<<<(512 * 512) / 256, 256, 0, stream>>>(pos, zero, neg, fragT);
  xpose_kernel<<<dim3(D_ / 32, N_ / 32, B_), dim3(32, 8), 0, stream>>>(x, xT);

  dim3 grid(D_ / BD, N_ / BT, B_);   // (2, 32, 8)
  toeplitz_wmma_kernel<<<grid, 256, 0, stream>>>(xT, fragT, out);
}
